// GCN_DGG_1984274891532
// MI455X (gfx1250) — matrix operand order repaired and checked by reading, not compiled, not run on hardware
//
#include <hip/hip_runtime.h>
#include <hip/hip_bf16.h>

// ---------------------------------------------------------------------------
// GCN forward, restructured for MI455X (gfx1250, wave32, WMMA):
//   d = rsqrt(rowsum(A)+1)
//   B1 = bf16(d ⊙ (x @ W1))                 [8192 x 256]
//   H  = bf16(relu(d ⊙ (A @ B1 + B1)))      [8192 x 256]   (self loop folded)
//   B2 = bf16(d ⊙ (H @ W2pad))              [8192 x 48]    (40 classes + pad)
//   Z2 = relu(d ⊙ (A @ B2 + B2))            [8192 x 48] f32
//   out = log_softmax(Z2[:, :40])           [8192 x 40] f32
// Heavy matmuls use v_wmma_f32_16x16x32_bf16 with fp32 accumulation.
// The adjacency operand is binary {0,1}: bf16 truncation is exact, so fp32->
// bf16 packing is done with a single v_perm_b32 per element pair.
// ---------------------------------------------------------------------------

#define NNODES 8192
#define NFEAT  512
#define NHID   256
#define NCLASS 40
#define NCPAD  48   // NCLASS padded to multiple of 16

typedef __attribute__((ext_vector_type(16))) __bf16 v16bf;
typedef __attribute__((ext_vector_type(8)))  float  v8f;

union FragAB {
    v16bf          v;
    unsigned short us[16];
    unsigned int   ui[8];
};

__device__ __forceinline__ unsigned short f2bf(float f) {
    unsigned int u = __float_as_uint(f);
    u += 0x7FFFu + ((u >> 16) & 1u);   // round-to-nearest-even
    return (unsigned short)(u >> 16);
}

__device__ __forceinline__ float bf2f(unsigned short h) {
    return __uint_as_float(((unsigned int)h) << 16);
}

// Pack two fp32 (given as raw bits) into a dword of two bf16 by truncation:
// result = { bf16(hi) , bf16(lo) }  ->  one v_perm_b32.
__device__ __forceinline__ unsigned int pack_bf16_trunc(unsigned int hi, unsigned int lo) {
    return __builtin_amdgcn_perm(hi, lo, 0x07060302u);
}

// --------------------------- degree kernel ---------------------------------
__global__ __launch_bounds__(256)
void degree_kernel(const float* __restrict__ adj, float* __restrict__ dvec, int n) {
    __shared__ float red[256];
    const float4* a = (const float4*)(adj + (size_t)blockIdx.x * n);
    float s = 0.f;
    for (int j = threadIdx.x; j < n / 4; j += 256) {
        float4 v = a[j];
        s += (v.x + v.y) + (v.z + v.w);
    }
    red[threadIdx.x] = s;
    __syncthreads();
    for (int off = 128; off > 0; off >>= 1) {
        if (threadIdx.x < off) red[threadIdx.x] += red[threadIdx.x + off];
        __syncthreads();
    }
    if (threadIdx.x == 0) dvec[blockIdx.x] = rsqrtf(red[0] + 1.0f);
}

// ----------------------- weight conversion kernels -------------------------
__global__ __launch_bounds__(256)
void cvt_f32_bf16_kernel(const float* __restrict__ src, unsigned short* __restrict__ dst, int n) {
    int i = blockIdx.x * blockDim.x + threadIdx.x;
    if (i < n) dst[i] = f2bf(src[i]);
}

__global__ __launch_bounds__(256)
void w2_pad_kernel(const float* __restrict__ W2, unsigned short* __restrict__ dst) {
    int i = blockIdx.x * blockDim.x + threadIdx.x;   // over NHID*NCPAD
    if (i >= NHID * NCPAD) return;
    int k = i / NCPAD, c = i % NCPAD;
    dst[i] = (c < NCLASS) ? f2bf(W2[k * NCLASS + c]) : (unsigned short)0;
}

// ----------------------------- WMMA GEMM -----------------------------------
// Computes, for each output element (m, n):
//   v = sum_k Acast[m,k] * B[k,n]           (bf16 x bf16 -> f32 accumulate)
//   if SELF: v += B[m,n]                    (adds the +I self-loop term)
//   v *= dvec[m]
//   if RELU: v = max(v, 0)
//   store f32 or bf16
// Wave layout per ISA: A fragment = lane holds row M=lane&15, K = 8h+s (s<8), 16+8h+s-8 (s>=8), h=lane>>4;
// B fragment = lane holds column N=lane&15, the same 16 K as A.
template<bool AF32, int NT, int WM, int WN, bool SELF, bool RELU, bool OUTF32>
__global__ __launch_bounds__(WM * WN * 32)
void wmma_gemm_kernel(const float* __restrict__ Af,
                      const unsigned short* __restrict__ Ab, int lda,
                      const unsigned short* __restrict__ B, int ldb,
                      void* __restrict__ outp, int ldo,
                      const float* __restrict__ dvec, int K) {
    const int lane = threadIdx.x & 31;
    const int wid  = threadIdx.x >> 5;
    const int wm   = wid % WM;
    const int wn   = wid / WM;
    const int m0   = (blockIdx.x * WM + wm) * 16;
    const int n0   = wn * (NT * 16);

    const v8f vzero = {0.f, 0.f, 0.f, 0.f, 0.f, 0.f, 0.f, 0.f};
    v8f acc[NT];
#pragma unroll
    for (int t = 0; t < NT; ++t) acc[t] = vzero;

    const int arow = m0 + (lane & 15);
    const int koff = (lane >> 4) << 3;    // upper 16 lanes take K+8

    for (int kc = 0; kc < K; kc += 32) {
        FragAB a;
        if constexpr (AF32) {
            const float* p = Af + (size_t)arow * lda + (kc + koff);
            // prefetch the streamed adjacency a few k-chunks ahead
            __builtin_prefetch(p + 128, 0, 0);
            uint4 u0 = *(const uint4*)(p);
            uint4 u1 = *(const uint4*)(p + 4);
            uint4 u2 = *(const uint4*)(p + 16);
            uint4 u3 = *(const uint4*)(p + 20);
            // truncating fp32->bf16 pack: exact for the binary adjacency
            a.ui[0] = pack_bf16_trunc(u0.y, u0.x);
            a.ui[1] = pack_bf16_trunc(u0.w, u0.z);
            a.ui[2] = pack_bf16_trunc(u1.y, u1.x);
            a.ui[3] = pack_bf16_trunc(u1.w, u1.z);
            a.ui[4] = pack_bf16_trunc(u2.y, u2.x);
            a.ui[5] = pack_bf16_trunc(u2.w, u2.z);
            a.ui[6] = pack_bf16_trunc(u3.y, u3.x);
            a.ui[7] = pack_bf16_trunc(u3.w, u3.z);
        } else {
            const uint4* p = (const uint4*)(Ab + (size_t)arow * lda + (kc + koff));
            const uint4* q = (const uint4*)(Ab + (size_t)arow * lda + (kc + koff + 16));
            uint4 u0 = *p, u1 = *q;
            a.ui[0] = u0.x; a.ui[1] = u0.y; a.ui[2] = u0.z; a.ui[3] = u0.w;
            a.ui[4] = u1.x; a.ui[5] = u1.y; a.ui[6] = u1.z; a.ui[7] = u1.w;
        }
#pragma unroll
        for (int t = 0; t < NT; ++t) {
            FragAB b;
            const unsigned short* bp = B + (size_t)(kc + koff) * ldb + (n0 + t * 16 + (lane & 15));
#pragma unroll
            for (int s = 0; s < 16; ++s) b.us[s] = bp[(size_t)(s < 8 ? s : s + 8) * ldb];
            acc[t] = __builtin_amdgcn_wmma_f32_16x16x32_bf16(
                false, a.v, false, b.v, (short)0, acc[t], false, false);
        }
    }

    // Epilogue. C layout: VGPR r, lane L -> (M = m0 + r + 8*(L>=16), N = n0 + L&15)
    const int mbase = m0 + ((lane >> 4) << 3);
    const int nlane = lane & 15;
#pragma unroll
    for (int t = 0; t < NT; ++t) {
        const int n = n0 + t * 16 + nlane;
#pragma unroll
        for (int r = 0; r < 8; ++r) {
            const int m = mbase + r;
            float v = acc[t][r];
            if constexpr (SELF) v += bf2f(B[(size_t)m * ldb + n]);
            v *= dvec[m];
            if constexpr (RELU) v = fmaxf(v, 0.f);
            if constexpr (OUTF32)
                ((float*)outp)[(size_t)m * ldo + n] = v;
            else
                ((unsigned short*)outp)[(size_t)m * ldo + n] = f2bf(v);
        }
    }
}

// ---------------------------- log_softmax ----------------------------------
__global__ __launch_bounds__(256)
void logsoftmax_kernel(const float* __restrict__ Z, float* __restrict__ out, int nrows) {
    int row = blockIdx.x * blockDim.x + threadIdx.x;
    if (row >= nrows) return;
    const float* z = Z + (size_t)row * NCPAD;
    float m = -1e30f;
#pragma unroll
    for (int c = 0; c < NCLASS; ++c) m = fmaxf(m, z[c]);
    float s = 0.f;
#pragma unroll
    for (int c = 0; c < NCLASS; ++c) s += __expf(z[c] - m);
    float lse = m + __logf(s);
#pragma unroll
    for (int c = 0; c < NCLASS; ++c) out[(size_t)row * NCLASS + c] = z[c] - lse;
}

// ------------------------------- launcher ----------------------------------
extern "C" void kernel_launch(void* const* d_in, const int* in_sizes, int n_in,
                              void* d_out, int out_size, void* d_ws, size_t ws_size,
                              hipStream_t stream) {
    const float* x   = (const float*)d_in[0];   // [8192 x 512]
    const float* adj = (const float*)d_in[1];   // [8192 x 8192]
    const float* W1  = (const float*)d_in[2];   // [512 x 256]
    const float* W2  = (const float*)d_in[3];   // [256 x 40]
    float* out = (float*)d_out;                 // [8192 x 40]

    char* ws = (char*)d_ws;
    size_t off = 0;
    auto alloc = [&](size_t bytes) -> void* {
        void* p = ws + off;
        off = (off + bytes + 255) & ~(size_t)255;
        return p;
    };
    float*          dvec = (float*)alloc((size_t)NNODES * 4);
    unsigned short* W1b  = (unsigned short*)alloc((size_t)NFEAT * NHID * 2);
    unsigned short* W2b  = (unsigned short*)alloc((size_t)NHID * NCPAD * 2);
    unsigned short* B1   = (unsigned short*)alloc((size_t)NNODES * NHID * 2);
    unsigned short* H    = (unsigned short*)alloc((size_t)NNODES * NHID * 2);
    unsigned short* B2   = (unsigned short*)alloc((size_t)NNODES * NCPAD * 2);
    float*          Z2   = (float*)alloc((size_t)NNODES * NCPAD * 4);

    // 1) degrees: d = rsqrt(rowsum(adj) + 1)
    degree_kernel<<<NNODES, 256, 0, stream>>>(adj, dvec, NNODES);

    // 2) weights to bf16 (W2 padded 40 -> 48 cols with zeros)
    cvt_f32_bf16_kernel<<<(NFEAT * NHID + 255) / 256, 256, 0, stream>>>(W1, W1b, NFEAT * NHID);
    w2_pad_kernel<<<(NHID * NCPAD + 255) / 256, 256, 0, stream>>>(W2, W2b);

    // 3) B1 = bf16(d ⊙ (x @ W1))          [8192 x 256], K = 512
    //    block = 8 waves (WM=4 x WN=2), wave = 16 rows x 8 tiles of 16 cols
    wmma_gemm_kernel<true, 8, 4, 2, false, false, false>
        <<<NNODES / 64, 256, 0, stream>>>(x, nullptr, NFEAT, W1b, NHID, B1, NHID, dvec, NFEAT);

    // 4) H = bf16(relu(d ⊙ (adj @ B1 + B1)))   [8192 x 256], K = 8192
    wmma_gemm_kernel<true, 8, 4, 2, true, true, false>
        <<<NNODES / 64, 256, 0, stream>>>(adj, nullptr, NNODES, B1, NHID, H, NHID, dvec, NNODES);

    // 5) B2 = bf16(d ⊙ (H @ W2b))         [8192 x 48], K = 256 (A already bf16)
    wmma_gemm_kernel<false, 3, 8, 1, false, false, false>
        <<<NNODES / 128, 256, 0, stream>>>(nullptr, H, NHID, W2b, NCPAD, B2, NCPAD, dvec, NHID);

    // 6) Z2 = relu(d ⊙ (adj @ B2 + B2))   [8192 x 48] f32, K = 8192
    wmma_gemm_kernel<true, 3, 8, 1, true, true, true>
        <<<NNODES / 128, 256, 0, stream>>>(adj, nullptr, NNODES, B2, NCPAD, Z2, NCPAD, dvec, NNODES);

    // 7) log_softmax over the 40 real classes
    logsoftmax_kernel<<<(NNODES + 255) / 256, 256, 0, stream>>>(Z2, out, NNODES);
}
